// SigLIPLoss_12713103196281
// MI455X (gfx1250) — compile-verified, matrix operand order checked by reading, not run on hardware
//
#include <hip/hip_runtime.h>
#include <hip/hip_bf16.h>

// ---------------------------------------------------------------------------
// SigLIP loss on gfx1250 (MI455X): normalize -> f16 -> WMMA GEMM with async
// global->LDS double-buffered staging and fused masked log-sigmoid epilogue.
// ---------------------------------------------------------------------------

typedef __attribute__((ext_vector_type(16))) _Float16 v16h;
typedef __attribute__((ext_vector_type(8)))  _Float16 v8h;
typedef __attribute__((ext_vector_type(4)))  _Float16 v4h;
typedef __attribute__((ext_vector_type(8)))  float    v8f;

#define GG 4
#define NN 4096
#define DD 1024
#define BT 128          // workgroup output tile (M and N)
#define KB 32           // k-chunk (one WMMA K step)
#define LDST 40         // LDS row stride in halfs (32 + 8 pad, keeps 16B align)

__device__ __constant__ float kMaxLogitScale = 4.605170185988091f;

// ---------------------------------------------------------------------------
// Kernel 0: zero the per-group accumulators (fresh every launch: deterministic)
// ---------------------------------------------------------------------------
__global__ void k_init(float* __restrict__ gacc) {
    if (threadIdx.x < GG) gacc[threadIdx.x] = 0.0f;
}

// ---------------------------------------------------------------------------
// Kernel 1: L2-normalize each row of merged/orig and store as f16.
// One wave (32 lanes) per row; each lane owns 32 floats (8 x float4).
// ---------------------------------------------------------------------------
__global__ void k_normalize(const float* __restrict__ merged,
                            const float* __restrict__ orig,
                            _Float16* __restrict__ zA,
                            _Float16* __restrict__ zB) {
    const int tid  = threadIdx.x;
    const int lane = tid & 31;
    const int wave = tid >> 5;
    const int r    = blockIdx.x * 8 + wave;      // 0 .. 2*GG*NN-1

    const float* src;
    _Float16*    dst;
    if (r < GG * NN) {
        src = merged + (size_t)r * DD;
        dst = zA + (size_t)r * DD;
    } else {
        src = orig + (size_t)(r - GG * NN) * DD;
        dst = zB + (size_t)(r - GG * NN) * DD;
    }

    float4 v[8];
    float  s = 0.0f;
#pragma unroll
    for (int c = 0; c < 8; ++c) {
        v[c] = *(const float4*)(src + c * 128 + lane * 4);
        s += v[c].x * v[c].x + v[c].y * v[c].y + v[c].z * v[c].z + v[c].w * v[c].w;
    }
#pragma unroll
    for (int off = 16; off > 0; off >>= 1) s += __shfl_xor(s, off, 32);

    const float inv = 1.0f / fmaxf(sqrtf(s), 1e-12f);

#pragma unroll
    for (int c = 0; c < 8; ++c) {
        v4h h;
        h[0] = (_Float16)(v[c].x * inv);
        h[1] = (_Float16)(v[c].y * inv);
        h[2] = (_Float16)(v[c].z * inv);
        h[3] = (_Float16)(v[c].w * inv);
        *(v4h*)(dst + c * 128 + lane * 4) = h;
    }
}

// ---------------------------------------------------------------------------
// Async copy of one 32-byte per-lane segment: two b128 global->LDS transfers.
// NOTE (ISA 08_async_tensor §4.4): INST_OFFSET is added to BOTH the LDS and
// the global address, so offset:16 covers the second 16B of a contiguous pair.
// ---------------------------------------------------------------------------
__device__ inline void async_cp_32B(unsigned ldsOff, const _Float16* gptr) {
    unsigned long long ga = (unsigned long long)(uintptr_t)gptr;
    asm volatile(
        "global_load_async_to_lds_b128 %0, %1, off\n\t"
        "global_load_async_to_lds_b128 %0, %1, off offset:16"
        :: "v"(ldsOff), "v"(ga) : "memory");
}

// ---------------------------------------------------------------------------
// Load one 16x32 f16 fragment from LDS in CDNA5 WMMA A/B register layout:
//   lanes 0-15 : row = lane,    halfs[0..7] = K 0..7,  halfs[8..15] = K 16..23
//   lanes 16-31: row = lane-16, halfs[0..7] = K 8..15, halfs[8..15] = K 24..31
// (B tile of logits = rows of z2, so A and B fragments load identically from
//  row-major [N, D] storage.)
// ---------------------------------------------------------------------------
__device__ inline v16h ld_frag(const _Float16* __restrict__ tile, int lane) {
    const int r  = lane & 15;
    const int k1 = (lane & 16) ? 8 : 0;
    const _Float16* p = tile + r * LDST;
    v8h lo = *(const v8h*)(p + k1);
    v8h hi = *(const v8h*)(p + k1 + 16);
    v16h o;
#pragma unroll
    for (int i = 0; i < 8; ++i) { o[i] = lo[i]; o[i + 8] = hi[i]; }
    return o;
}

// ---------------------------------------------------------------------------
// Kernel 2: tiled GEMM (logits = z1 . z2^T per group) with fused masked
// log-sigmoid loss epilogue. 256 threads = 8 waves; each wave computes a
// 64x32 region = 4x2 WMMA accumulators. Double-buffered async LDS staging.
// ---------------------------------------------------------------------------
__global__ __launch_bounds__(256)
void k_siglip_gemm(const _Float16* __restrict__ zA,
                   const _Float16* __restrict__ zB,
                   const unsigned char* __restrict__ masks,
                   const float* __restrict__ logit_scale,
                   const float* __restrict__ logit_bias,
                   float* __restrict__ gacc) {
    __shared__ _Float16 As[2][BT * LDST];
    __shared__ _Float16 Bs[2][BT * LDST];

    const int tid   = threadIdx.x;
    const int lane  = tid & 31;
    const int wave  = tid >> 5;
    const int g     = blockIdx.z;
    const int row0  = blockIdx.y * BT;   // z1 rows (M)
    const int col0  = blockIdx.x * BT;   // z2 rows (N of logits)
    const int waveM = wave >> 2;         // 0..1 -> 64-row strip
    const int waveN = wave & 3;          // 0..3 -> 32-col strip

    const _Float16* Ag = zA + (size_t)g * NN * DD;
    const _Float16* Bg = zB + (size_t)g * NN * DD;

    // cooperative loader: thread -> (row, 16-half segment)
    const int lrow = tid >> 1;
    const int lseg = (tid & 1) * 16;
    const _Float16* gA = Ag + (size_t)(row0 + lrow) * DD + lseg;
    const _Float16* gB = Bg + (size_t)(col0 + lrow) * DD + lseg;

    // LDS byte offsets for the async engine (aperture rule: addr[31:0] = LDS offset)
    unsigned sAoff[2], sBoff[2];
    sAoff[0] = (unsigned)(uintptr_t)(const void*)&As[0][lrow * LDST + lseg];
    sAoff[1] = (unsigned)(uintptr_t)(const void*)&As[1][lrow * LDST + lseg];
    sBoff[0] = (unsigned)(uintptr_t)(const void*)&Bs[0][lrow * LDST + lseg];
    sBoff[1] = (unsigned)(uintptr_t)(const void*)&Bs[1][lrow * LDST + lseg];

    v8f acc[4][2];
#pragma unroll
    for (int i = 0; i < 4; ++i)
#pragma unroll
        for (int j = 0; j < 2; ++j)
#pragma unroll
            for (int k = 0; k < 8; ++k) acc[i][j][k] = 0.0f;

    const int nChunks = DD / KB;   // 32

    // prologue: fill buffer 0 with chunk 0 (4 async issues per wave)
    async_cp_32B(sAoff[0], gA);
    async_cp_32B(sBoff[0], gB);

    for (int ic = 0; ic < nChunks; ++ic) {
        const int buf = ic & 1;
        if (ic + 1 < nChunks) {
            // kick off next chunk into the other buffer, then wait until only
            // those 4 new issues are outstanding (async loads complete in order)
            async_cp_32B(sAoff[buf ^ 1], gA + (ic + 1) * KB);
            async_cp_32B(sBoff[buf ^ 1], gB + (ic + 1) * KB);
            asm volatile("s_wait_asynccnt 0x4" ::: "memory");
        } else {
            asm volatile("s_wait_asynccnt 0x0" ::: "memory");
        }
        __syncthreads();   // current buffer visible to all waves

        v16h a[4], b[2];
#pragma unroll
        for (int mt = 0; mt < 4; ++mt)
            a[mt] = ld_frag(&As[buf][(waveM * 64 + mt * 16) * LDST], lane);
#pragma unroll
        for (int nt = 0; nt < 2; ++nt)
            b[nt] = ld_frag(&Bs[buf][(waveN * 32 + nt * 16) * LDST], lane);

#pragma unroll
        for (int mt = 0; mt < 4; ++mt)
#pragma unroll
            for (int nt = 0; nt < 2; ++nt)
                acc[mt][nt] = __builtin_amdgcn_wmma_f32_16x16x32_f16(
                    /*neg_a=*/false, a[mt], /*neg_b=*/false, b[nt],
                    /*c_mod=*/(short)0, acc[mt][nt],
                    /*reuse_a=*/false, /*reuse_b=*/false);

        __syncthreads();   // everyone done reading buf before it is refilled
    }

    // ---- fused epilogue: x = c*scale + bias; loss -= w * logsig(label*x) ----
    const float scale = __expf(fminf(logit_scale[0], kMaxLogitScale));
    const float bias  = logit_bias[0];
    const unsigned char* mg = masks + (size_t)g * NN;

    // C/D layout: N = lane&15 (+16 per nt tile), M = vgpr + 8*(lane>>4)
    const int nBase = col0 + waveN * 32 + (lane & 15);
    const int mBase = row0 + waveM * 64 + ((lane >> 4) << 3);

    float local = 0.0f;
#pragma unroll
    for (int mt = 0; mt < 4; ++mt) {
#pragma unroll
        for (int nt = 0; nt < 2; ++nt) {
            const int   nIdx = nBase + nt * 16;
            const float wN   = mg[nIdx] ? 1.0f : 0.0f;
#pragma unroll
            for (int v = 0; v < 8; ++v) {
                const int   mIdx = mBase + mt * 16 + v;
                const float x    = acc[mt][nt][v] * scale + bias;
                const float t    = (mIdx == nIdx) ? x : -x;   // label * logit
                // stable log_sigmoid(t) = min(t,0) - log1p(exp(-|t|))
                const float lsg  = fminf(t, 0.0f) - log1pf(__expf(-fabsf(t)));
                const float wM   = mg[mIdx] ? 1.0f : 0.0f;
                local -= wN * wM * lsg;   // accumulate POSITIVE loss
            }
        }
    }

#pragma unroll
    for (int off = 16; off > 0; off >>= 1) local += __shfl_down(local, off, 32);
    if (lane == 0) atomicAdd(&gacc[g], local);
}

// ---------------------------------------------------------------------------
// Kernel 3: per-group normalization by n_sel^2 and mean over valid groups.
// ---------------------------------------------------------------------------
__global__ void k_finalize(const unsigned char* __restrict__ masks,
                           const float* __restrict__ gacc,
                           float* __restrict__ out) {
    __shared__ float pg[GG];
    __shared__ int   valid[GG];

    const int g    = threadIdx.x >> 5;   // blockDim.x == 128 -> g in 0..3
    const int lane = threadIdx.x & 31;

    float cnt = 0.0f;
    for (int i = lane; i < NN; i += 32) cnt += masks[(size_t)g * NN + i] ? 1.0f : 0.0f;
#pragma unroll
    for (int off = 16; off > 0; off >>= 1) cnt += __shfl_xor(cnt, off, 32);

    if (lane == 0) {
        const float denom = fmaxf(cnt * cnt, 1.0f);
        pg[g]    = gacc[g] / denom;
        valid[g] = (cnt > 0.0f) ? 1 : 0;
    }
    __syncthreads();

    if (threadIdx.x == 0) {
        float sum = 0.0f, nv = 0.0f;
        for (int i = 0; i < GG; ++i) {
            if (valid[i]) { sum += pg[i]; nv += 1.0f; }
        }
        out[0] = sum / fmaxf(nv, 1.0f);
    }
}

// ---------------------------------------------------------------------------
// Host-side launcher
// ---------------------------------------------------------------------------
extern "C" void kernel_launch(void* const* d_in, const int* in_sizes, int n_in,
                              void* d_out, int out_size, void* d_ws, size_t ws_size,
                              hipStream_t stream) {
    const float*         merged = (const float*)d_in[0];
    const float*         orig   = (const float*)d_in[1];
    const unsigned char* masks  = (const unsigned char*)d_in[2];
    const float*         lscale = (const float*)d_in[3];
    const float*         lbias  = (const float*)d_in[4];
    float*               out    = (float*)d_out;

    // Workspace layout: zA (32 MB f16) | zB (32 MB f16) | gacc (4 floats)
    char*      ws  = (char*)d_ws;
    _Float16*  zA  = (_Float16*)ws;
    _Float16*  zB  = (_Float16*)(ws + (size_t)GG * NN * DD * sizeof(_Float16));
    float*    gacc = (float*)(ws + (size_t)GG * NN * DD * sizeof(_Float16) * 2);

    k_init<<<1, 32, 0, stream>>>(gacc);

    // 2*G*N rows, 8 waves (rows) per 256-thread block
    k_normalize<<<(2 * GG * NN) / 8, 256, 0, stream>>>(merged, orig, zA, zB);

    dim3 grid(NN / BT, NN / BT, GG);
    k_siglip_gemm<<<grid, 256, 0, stream>>>(zA, zB, masks, lscale, lbias, gacc);

    k_finalize<<<1, 128, 0, stream>>>(masks, gacc, out);
}
